// AdaptiveAttention_68315749810847
// MI455X (gfx1250) — matrix operand order repaired and checked by reading, not compiled, not run on hardware
//
#include <hip/hip_runtime.h>

#define SEQ 2048
#define DM  1024
#define NH  16
#define DK  64

typedef __attribute__((ext_vector_type(16))) __bf16 bf16x16;
typedef __attribute__((ext_vector_type(8)))  float  f32x8;
typedef unsigned int u32;
typedef __attribute__((ext_vector_type(4))) u32 u32x4;
typedef __attribute__((ext_vector_type(8))) int i32x8;
typedef __attribute__((ext_vector_type(4))) int i32x4;

#if defined(__HIP_DEVICE_COMPILE__) && __has_builtin(__builtin_amdgcn_tensor_load_to_lds)
#define HAVE_TDM 1
#else
#define HAVE_TDM 0
#endif

#if __has_include(<hip/amd_detail/amd_gfx1250_TDM.h>)
#define TDM_SIX_ARGS 1
#else
#define TDM_SIX_ARGS 0
#endif

// ---------- bf16 helpers (round-to-nearest-even) ----------
__device__ inline unsigned short f2bf_bits(float f) {
  unsigned u = __builtin_bit_cast(unsigned, f);
  u += 0x7FFFu + ((u >> 16) & 1u);
  return (unsigned short)(u >> 16);
}
__device__ inline __bf16 f2bf(float f) {
  unsigned short h = f2bf_bits(f);
  return __builtin_bit_cast(__bf16, h);
}

// ---------- WMMA fragment helpers (CDNA5 16x16x32 bf16 layouts) ----------
union FragU { bf16x16 v; uint4 q[2]; };

// A fragment (16x32): lane r = row r; VGPRs 0-3 = K {k0+half*8..+7},
// VGPRs 4-7 = K {k0+16+half*8..+7}. base row-major, stride in elements.
__device__ inline bf16x16 load_a(const __bf16* base, int stride, int r, int half, int k0) {
  FragU f;
  const __bf16* p = base + r * stride + k0 + half * 8;
  f.q[0] = *(const uint4*)(p);
  f.q[1] = *(const uint4*)(p + 16);
  return f.v;
}
// B fragment (32x16): lane r = output column `row`; VGPRs 0-3 = K {k0+half*8..+7},
// VGPRs 4-7 = K {k0+16+half*8..+7}.
// bt is stored [col][K] contiguous (B transposed), stride in elements.
__device__ inline bf16x16 load_b(const __bf16* bt, int stride, int row, int half, int k0) {
  FragU f;
  const __bf16* p = bt + row * stride + k0 + half * 8;
  f.q[0] = *(const uint4*)(p);
  f.q[1] = *(const uint4*)(p + 16);
  return f.v;
}
__device__ inline f32x8 wmma_bf16(bf16x16 a, bf16x16 b, f32x8 c) {
  return __builtin_amdgcn_wmma_f32_16x16x32_bf16(false, a, false, b, (short)0, c,
                                                 false, false);
}

// ---------- Tensor Data Mover: 2-D bf16 tile -> LDS with row padding ----------
// Generic LDS pointer low 32 bits == LDS byte offset (flat aperture rule).
__device__ inline unsigned lds_off(const void* p) {
  return (unsigned)(unsigned long long)p;
}

#if HAVE_TDM
// D# packed per CDNA5 ISA ch.8: group0 {count=1, lds_addr, global_addr, type=2};
// group1 {data_size=2B, pad_enable, pad_interval(2<<c DW), pad_amount(c+1 DW),
//         tensor_dim0/1, tile_dim0/1, tensor_dim0_stride}.  Dims in elements.
__device__ inline void tdm_load_2d(unsigned lds_byte, const void* gptr,
                                   u32 tensor_d0, u32 tensor_d1,
                                   unsigned long long stride_d0_elems,
                                   u32 tile_d0, u32 tile_d1,
                                   u32 pad_int_code, u32 pad_amt_code) {
  const unsigned long long ga = (unsigned long long)gptr;
  u32x4 g0;
  g0[0] = 1u;                                               // count=1, user desc
  g0[1] = lds_byte;                                         // lds_addr
  g0[2] = (u32)ga;                                          // global_addr lo
  g0[3] = (u32)((ga >> 32) & 0x01FFFFFFu) | (2u << 30);     // ga hi | type=2
  i32x8 g1;
  g1[0] = (int)((1u << 16) | (1u << 20) |                   // data_size=2B, pad_en
                (pad_int_code << 22) | (pad_amt_code << 25));
  g1[1] = (int)((tensor_d0 & 0xFFFFu) << 16);               // tensor_dim0[15:0]
  g1[2] = (int)((tensor_d0 >> 16) | ((tensor_d1 & 0xFFFFu) << 16));
  g1[3] = (int)((tensor_d1 >> 16) | (tile_d0 << 16));       // tile_dim0
  g1[4] = (int)(tile_d1 & 0xFFFFu);                         // tile_dim1, tile_dim2=0
  g1[5] = (int)(u32)(stride_d0_elems & 0xFFFFFFFFu);        // dim0 stride lo
  g1[6] = (int)((stride_d0_elems >> 32) & 0xFFFFu);         // stride hi, dim1 stride=0
  g1[7] = 0;
  const i32x4 z4 = (i32x4)0;
#if TDM_SIX_ARGS
  __builtin_amdgcn_tensor_load_to_lds(g0, g1, z4, z4, (i32x8)0, 0);
#else
  __builtin_amdgcn_tensor_load_to_lds(g0, g1, z4, z4, 0);
#endif
}
#endif

// =====================================================================
// Streaming f32 -> bf16 convert (8 elements / thread)
// =====================================================================
__global__ __launch_bounds__(256) void cvt_kernel(const float* __restrict__ s,
                                                  __bf16* __restrict__ d) {
  const size_t i = ((size_t)blockIdx.x * 256 + threadIdx.x) * 8;
  const float4 a = *(const float4*)(s + i);
  const float4 b = *(const float4*)(s + i + 4);
  uint4 o;
  o.x = (u32)f2bf_bits(a.x) | ((u32)f2bf_bits(a.y) << 16);
  o.y = (u32)f2bf_bits(a.z) | ((u32)f2bf_bits(a.w) << 16);
  o.z = (u32)f2bf_bits(b.x) | ((u32)f2bf_bits(b.y) << 16);
  o.w = (u32)f2bf_bits(b.z) | ((u32)f2bf_bits(b.w) << 16);
  *(uint4*)(d + i) = o;
}

// =====================================================================
// Weight transpose + convert: Wt[n][k] (bf16) = W[k][n] (f32), 32x32 LDS tiles
// =====================================================================
__global__ __launch_bounds__(256) void transpose_cvt_kernel(const float* __restrict__ W,
                                                            __bf16* __restrict__ Wt) {
  __shared__ __bf16 T[32 * 33];
  const int k0 = blockIdx.y * 32, n0 = blockIdx.x * 32;
  const int tid = threadIdx.x;
  {
    const int i = tid >> 3, j4 = (tid & 7) << 2;
    const float4 v = *(const float4*)(W + (size_t)(k0 + i) * DM + n0 + j4);
    T[(j4 + 0) * 33 + i] = f2bf(v.x);
    T[(j4 + 1) * 33 + i] = f2bf(v.y);
    T[(j4 + 2) * 33 + i] = f2bf(v.z);
    T[(j4 + 3) * 33 + i] = f2bf(v.w);
  }
  __syncthreads();
  {
    const int j = tid >> 3, k4 = (tid & 7) << 2;
    uint2 pk;
    pk.x = (u32)__builtin_bit_cast(unsigned short, T[j * 33 + k4 + 0]) |
           ((u32)__builtin_bit_cast(unsigned short, T[j * 33 + k4 + 1]) << 16);
    pk.y = (u32)__builtin_bit_cast(unsigned short, T[j * 33 + k4 + 2]) |
           ((u32)__builtin_bit_cast(unsigned short, T[j * 33 + k4 + 3]) << 16);
    *(uint2*)(Wt + (size_t)(n0 + j) * DM + k0 + k4) = pk;
  }
}

// =====================================================================
// GEMM: C[M,N] = (A[M,K](bf16) @ Bt[N,K]^T(bf16) + bias) * scale,
// double-buffered TDM staging.  Block tile 128(M) x 64(N), K-step 32,
// 8 waves (one 16-row subtile each).
// MODE 0: bf16 out in [B,H,S,dk] head layout;  MODE 1: f32 linear [M,N].
// scale folds dk^-0.5 into the Q projection (softmax((QK^T)s)=softmax((Qs)K^T)).
// =====================================================================
template <int MODE>
__global__ __launch_bounds__(256) void gemm_kernel(const __bf16* __restrict__ A,
                                                   const __bf16* __restrict__ Bt,
                                                   const float* __restrict__ bias,
                                                   void* __restrict__ out,
                                                   int M, int K, int N, float scale) {
  // padded stride 40 elems = 80 B: 16B-aligned rows, conflict-free b128 reads
  __shared__ __align__(16) __bf16 As[2][128 * 40];
  __shared__ __align__(16) __bf16 Bs[2][64 * 40];

  const int tid = threadIdx.x, wid = tid >> 5, lane = tid & 31;
  const int half = lane >> 4, r = lane & 15;
  const int mbase = blockIdx.y * 128, nbase = blockIdx.x * 64;
  const int niter = K / 32;

  auto stage = [&](int it, int buf) {
    const int kb = it * 32;
#if HAVE_TDM
    if (wid == 0) {
      // A tile: 128 rows x 32 k; 64B rows (16 DW -> code 3) + 16B pad (code 3)
      tdm_load_2d(lds_off(&As[buf][0]), A + (size_t)mbase * K + kb,
                  (u32)(K - kb), (u32)(M - mbase), (unsigned long long)K,
                  32u, 128u, 3u, 3u);
      // B tile: 64 n-rows x 32 k from pre-transposed weights
      tdm_load_2d(lds_off(&Bs[buf][0]), Bt + (size_t)nbase * K + kb,
                  (u32)(K - kb), (u32)(N - nbase), (unsigned long long)K,
                  32u, 64u, 3u, 3u);
    }
#else
    for (int c = tid; c < 512; c += 256) {
      const int row = c >> 2, c8 = (c & 3) << 3;
      *(uint4*)(&As[buf][row * 40 + c8]) =
          *(const uint4*)(A + (size_t)(mbase + row) * K + kb + c8);
    }
    for (int c = tid; c < 256; c += 256) {
      const int row = c >> 2, c8 = (c & 3) << 3;
      *(uint4*)(&Bs[buf][row * 40 + c8]) =
          *(const uint4*)(Bt + (size_t)(nbase + row) * K + kb + c8);
    }
#endif
  };

  f32x8 acc[4];
#pragma unroll
  for (int t = 0; t < 4; ++t) acc[t] = (f32x8)0.0f;

  stage(0, 0);
  for (int it = 0; it < niter; ++it) {
    const int buf = it & 1;
#if HAVE_TDM
    if (wid == 0) __builtin_amdgcn_s_wait_tensorcnt(0);
#endif
    __syncthreads();
    if (it + 1 < niter) stage(it + 1, buf ^ 1);

    const bf16x16 af = load_a(&As[buf][wid * 16 * 40], 40, r, half, 0);
#pragma unroll
    for (int t = 0; t < 4; ++t) {
      const bf16x16 bf = load_b(&Bs[buf][0], 40, t * 16 + r, half, 0);
      acc[t] = wmma_bf16(af, bf, acc[t]);
    }
  }

#pragma unroll
  for (int t = 0; t < 4; ++t) {
    const int n = nbase + t * 16 + r;
    const float bv = bias[n];
#pragma unroll
    for (int i = 0; i < 8; ++i) {
      const int m = mbase + wid * 16 + i + half * 8;
      const float v = (acc[t][i] + bv) * scale;
      if (MODE == 0) {
        const int b = m >> 11, s = m & (SEQ - 1);
        const int h = n >> 6, d = n & (DK - 1);
        ((__bf16*)out)[(((size_t)(b * NH + h) * SEQ + s) * DK) + d] = f2bf(v);
      } else {
        ((float*)out)[(size_t)m * N + n] = v;
      }
    }
  }
}

// =====================================================================
// gate[b,h,s] = sigmoid(query[b,s,:] . Wg[:,h] + bg[h])   (67 MFLOP, trivial)
// =====================================================================
__global__ __launch_bounds__(256) void gate_kernel(const float* __restrict__ query,
                                                   const float* __restrict__ Wg,
                                                   const float* __restrict__ bg,
                                                   float* __restrict__ gate) {
  const int idx = blockIdx.x * 256 + threadIdx.x;  // B*S*H
  const int m = idx >> 4, h = idx & 15;
  const int b = m >> 11, s = m & (SEQ - 1);
  const float* q = query + (size_t)m * DM;
  float acc = bg[h];
  for (int k = 0; k < DM; ++k) acc += q[k] * Wg[k * NH + h];
  gate[(size_t)(b * NH + h) * SEQ + s] = 1.0f / (1.0f + __expf(-acc));
}

// =====================================================================
// Flash attention, double-buffered: K tiles via TDM, V tiles via vector
// loads with transpose; online softmax in registers; gate folded into 1/l.
// Q already carries the dk^-0.5 scale from its projection.
// =====================================================================
__global__ __launch_bounds__(256) void attn_kernel(const __bf16* __restrict__ Qbf,
                                                   const __bf16* __restrict__ Kbf,
                                                   const __bf16* __restrict__ Vbf,
                                                   const float* __restrict__ gate,
                                                   __bf16* __restrict__ ctx) {
  __shared__ __align__(16) __bf16 Kt[2][64 * 72];  // [key][dk], stride 72
  __shared__ __align__(16) __bf16 Vt[2][64 * 72];  // transposed [dk][key]
  __shared__ __align__(16) __bf16 Pb[8 * 16 * 72]; // per-wave P relayout

  const int tid = threadIdx.x, wid = tid >> 5, lane = tid & 31;
  const int half = lane >> 4, r = lane & 15;
  const int bh = blockIdx.y, b = bh >> 4, h = bh & 15;
  const int q0 = blockIdx.x * 128 + wid * 16;

  const __bf16* Qrow = Qbf + ((size_t)bh * SEQ + q0) * DK;
  const bf16x16 aq0 = load_a(Qrow, DK, r, half, 0);
  const bf16x16 aq1 = load_a(Qrow, DK, r, half, 32);

  auto stageK = [&](int i, int buf) {
    const int kt = i * 64;
#if HAVE_TDM
    if (wid == 0)  // 128B rows (32 DW -> code 4) + 16B pad (code 3) -> stride 72
      tdm_load_2d(lds_off(&Kt[buf][0]), Kbf + ((size_t)bh * SEQ + kt) * DK,
                  (u32)DK, (u32)(SEQ - kt), (unsigned long long)DK, 64u, 64u, 4u, 3u);
#else
    const __bf16* Ksrc = Kbf + ((size_t)bh * SEQ + kt) * DK;
    for (int c = tid; c < 512; c += 256) {
      const int row = c >> 3, c8 = (c & 7) << 3;
      *(uint4*)(&Kt[buf][row * 72 + c8]) = *(const uint4*)(Ksrc + row * DK + c8);
    }
#endif
  };
  auto stageV = [&](int i, int buf) {
    const __bf16* Vsrc = Vbf + ((size_t)bh * SEQ + i * 64) * DK;
    for (int c = tid; c < 512; c += 256) {
      const int row = c >> 3, c8 = (c & 7) << 3;
      const uint4 v = *(const uint4*)(Vsrc + row * DK + c8);
      const __bf16* e = (const __bf16*)&v;
#pragma unroll
      for (int j = 0; j < 8; ++j) Vt[buf][(c8 + j) * 72 + row] = e[j];
    }
  };

  f32x8 O[4];
#pragma unroll
  for (int t = 0; t < 4; ++t) O[t] = (f32x8)0.0f;
  float mrow[8], lrow[8];
#pragma unroll
  for (int i = 0; i < 8; ++i) { mrow[i] = -1e30f; lrow[i] = 0.0f; }

  const int nkt = SEQ / 64;
  stageK(0, 0);
  stageV(0, 0);
  for (int i = 0; i < nkt; ++i) {
    const int buf = i & 1;
#if HAVE_TDM
    if (wid == 0) __builtin_amdgcn_s_wait_tensorcnt(0);
#endif
    __syncthreads();
    if (i + 1 < nkt) { stageK(i + 1, buf ^ 1); stageV(i + 1, buf ^ 1); }

    // ---- scores: S = Q K^T (Q pre-scaled), 16 x 64 per wave ----
    f32x8 sc[4];
#pragma unroll
    for (int t = 0; t < 4; ++t) {
      const bf16x16 b0 = load_b(&Kt[buf][0], 72, t * 16 + r, half, 0);
      f32x8 s = wmma_bf16(aq0, b0, (f32x8)0.0f);
      const bf16x16 b1 = load_b(&Kt[buf][0], 72, t * 16 + r, half, 32);
      sc[t] = wmma_bf16(aq1, b1, s);
    }

    // ---- online softmax (row reductions inside 16-lane halves) ----
    float mnew[8], alpha[8], rs[8];
#pragma unroll
    for (int i2 = 0; i2 < 8; ++i2) {
      float mx = fmaxf(fmaxf(sc[0][i2], sc[1][i2]), fmaxf(sc[2][i2], sc[3][i2]));
#pragma unroll
      for (int d = 1; d < 16; d <<= 1) mx = fmaxf(mx, __shfl_xor(mx, d, 32));
      mnew[i2] = fmaxf(mrow[i2], mx);
      alpha[i2] = __expf(mrow[i2] - mnew[i2]);
      rs[i2] = 0.0f;
    }
#pragma unroll
    for (int t = 0; t < 4; ++t)
#pragma unroll
      for (int i2 = 0; i2 < 8; ++i2) {
        const float p = __expf(sc[t][i2] - mnew[i2]);
        sc[t][i2] = p;
        rs[i2] += p;
      }
#pragma unroll
    for (int i2 = 0; i2 < 8; ++i2) {
      float s = rs[i2];
#pragma unroll
      for (int d = 1; d < 16; d <<= 1) s += __shfl_xor(s, d, 32);
      lrow[i2] = lrow[i2] * alpha[i2] + s;
      mrow[i2] = mnew[i2];
    }

    // ---- relayout P (C-layout) -> A-layout via wave-private LDS ----
    __bf16* P = Pb + wid * 16 * 72;
#pragma unroll
    for (int t = 0; t < 4; ++t)
#pragma unroll
      for (int i2 = 0; i2 < 8; ++i2)
        P[(i2 + half * 8) * 72 + t * 16 + r] = f2bf(sc[t][i2]);

    const bf16x16 pa0 = load_a(P, 72, r, half, 0);
    const bf16x16 pa1 = load_a(P, 72, r, half, 32);

    // ---- O = O*alpha + P @ V ----
#pragma unroll
    for (int t = 0; t < 4; ++t) {
      f32x8 o = O[t];
#pragma unroll
      for (int i2 = 0; i2 < 8; ++i2) o[i2] *= alpha[i2];
      const bf16x16 bv0 = load_b(&Vt[buf][0], 72, t * 16 + r, half, 0);
      o = wmma_bf16(pa0, bv0, o);
      const bf16x16 bv1 = load_b(&Vt[buf][0], 72, t * 16 + r, half, 32);
      O[t] = wmma_bf16(pa1, bv1, o);
    }
  }

  // ---- finalize: ctx = gate * O / l, bf16 in [B,S,D] layout ----
  float gscale[8];
#pragma unroll
  for (int i = 0; i < 8; ++i) {
    const int q = q0 + i + half * 8;
    gscale[i] = gate[(size_t)bh * SEQ + q] / lrow[i];
  }
#pragma unroll
  for (int t = 0; t < 4; ++t)
#pragma unroll
    for (int i = 0; i < 8; ++i) {
      const int q = q0 + i + half * 8;
      const int n = t * 16 + r;
      ctx[((size_t)(b * SEQ + q)) * DM + h * DK + n] = f2bf(O[t][i] * gscale[i]);
    }
}

// =====================================================================
extern "C" void kernel_launch(void* const* d_in, const int* in_sizes, int n_in,
                              void* d_out, int out_size, void* d_ws, size_t ws_size,
                              hipStream_t stream) {
  (void)in_sizes; (void)n_in; (void)out_size; (void)ws_size;

  const float* query = (const float*)d_in[0];
  const float* key_  = (const float*)d_in[1];
  const float* value = (const float*)d_in[2];
  const float* Wq = (const float*)d_in[3];
  const float* bq = (const float*)d_in[4];
  const float* Wk = (const float*)d_in[5];
  const float* bk = (const float*)d_in[6];
  const float* Wv = (const float*)d_in[7];
  const float* bv = (const float*)d_in[8];
  const float* Wo = (const float*)d_in[9];
  const float* bo = (const float*)d_in[10];
  const float* Wg = (const float*)d_in[11];
  const float* bg = (const float*)d_in[12];

  const size_t ACT = (size_t)2 * SEQ * DM;  // 4 Mi elements
  const size_t WEL = (size_t)DM * DM;       // 1 Mi elements
  __bf16* p = (__bf16*)d_ws;
  __bf16* qx = p; p += ACT;
  __bf16* kx = p; p += ACT;
  __bf16* vx = p; p += ACT;
  __bf16* Wtq = p; p += WEL;
  __bf16* Wtk = p; p += WEL;
  __bf16* Wtv = p; p += WEL;
  __bf16* Wto = p; p += WEL;
  __bf16* Qbf = p; p += ACT;
  __bf16* Kbf = p; p += ACT;
  __bf16* Vbf = p; p += ACT;
  __bf16* ctxbf = p; p += ACT;
  float* gate = (float*)p;

  const dim3 blk(256);
  const dim3 gcv(ACT / (256 * 8));             // 2048 blocks
  const dim3 gtr(32, 32);
  const dim3 gproj(DM / 64, (2 * SEQ) / 128);  // (16, 32)

  hipLaunchKernelGGL(cvt_kernel, gcv, blk, 0, stream, query, qx);
  hipLaunchKernelGGL(cvt_kernel, gcv, blk, 0, stream, key_, kx);
  hipLaunchKernelGGL(cvt_kernel, gcv, blk, 0, stream, value, vx);
  hipLaunchKernelGGL(transpose_cvt_kernel, gtr, blk, 0, stream, Wq, Wtq);
  hipLaunchKernelGGL(transpose_cvt_kernel, gtr, blk, 0, stream, Wk, Wtk);
  hipLaunchKernelGGL(transpose_cvt_kernel, gtr, blk, 0, stream, Wv, Wtv);
  hipLaunchKernelGGL(transpose_cvt_kernel, gtr, blk, 0, stream, Wo, Wto);

  // Q projection carries dk^-0.5 (= 0.125) so the attention loop skips scaling.
  hipLaunchKernelGGL((gemm_kernel<0>), gproj, blk, 0, stream,
                     qx, Wtq, bq, (void*)Qbf, 2 * SEQ, DM, DM, 0.125f);
  hipLaunchKernelGGL((gemm_kernel<0>), gproj, blk, 0, stream,
                     kx, Wtk, bk, (void*)Kbf, 2 * SEQ, DM, DM, 1.0f);
  hipLaunchKernelGGL((gemm_kernel<0>), gproj, blk, 0, stream,
                     vx, Wtv, bv, (void*)Vbf, 2 * SEQ, DM, DM, 1.0f);

  hipLaunchKernelGGL(gate_kernel, dim3((2 * SEQ * NH) / 256), blk, 0, stream,
                     query, Wg, bg, gate);

  hipLaunchKernelGGL(attn_kernel, dim3(SEQ / 128, 2 * NH), blk, 0, stream,
                     Qbf, Kbf, Vbf, gate, ctxbf);

  hipLaunchKernelGGL((gemm_kernel<1>), gproj, blk, 0, stream,
                     ctxbf, Wto, bo, d_out, 2 * SEQ, DM, DM, 1.0f);
}
